// GCNEdgeClassifier_40037685133539
// MI455X (gfx1250) — matrix operand order repaired and checked by reading, not compiled, not run on hardware
//
#include <hip/hip_runtime.h>

#define N_NODES 50000
#define IN_C    256
#define HID     128
#define NT      (HID / 16)          // 8 column tiles
#define CHUNK   (NT * 32 * 16)      // 4096 bf16 = 8 KB packed-W per 32-K chunk

typedef __attribute__((ext_vector_type(16))) __bf16 v16bf;
typedef __attribute__((ext_vector_type(8)))  __bf16 v8bf;
typedef __attribute__((ext_vector_type(8)))  float  v8f;

// f32 -> bf16 round-to-nearest-even
__device__ __forceinline__ __bf16 f2bf(float f) {
    unsigned u = __builtin_bit_cast(unsigned, f);
    u += 0x7FFFu + ((u >> 16) & 1u);
    unsigned short h = (unsigned short)(u >> 16);
    return __builtin_bit_cast(__bf16, h);
}

__device__ __forceinline__ v16bf cat16(v8bf lo, v8bf hi) {
    return __builtin_shufflevector(lo, hi, 0, 1, 2, 3, 4, 5, 6, 7,
                                           8, 9, 10, 11, 12, 13, 14, 15);
}

// elementwise f32 -> bf16
__global__ void cvt_bf16(const float* __restrict__ in, __bf16* __restrict__ out, int n) {
    int i = blockIdx.x * blockDim.x + threadIdx.x;
    if (i < n) out[i] = f2bf(in[i]);
}

// Pre-swizzle W[K,HID] (f32 row-major) into per-lane WMMA B-fragment layout (bf16):
// Wp[((kt*NT+t)*32 + lane)*16 + i] = bf16(W[kt*32 + (lane>>4)*16 + i][t*16 + (lane&15)])
__global__ void pack_weight(const float* __restrict__ W, __bf16* __restrict__ Wp, int K) {
    int idx = blockIdx.x * blockDim.x + threadIdx.x;
    if (idx >= K * HID) return;
    int i    = idx & 15;
    int lane = (idx >> 4) & 31;
    int frag = idx >> 9;            // kt*NT + t
    int t    = frag & (NT - 1);
    int kt   = frag / NT;
    int k = kt * 32 + (lane >> 4) * 16 + i;
    int n = t * 16 + (lane & 15);
    Wp[idx] = f2bf(W[(size_t)k * HID + n]);
}

// out[M,HID] = A[M,K](bf16) @ W(packed bf16), f32 accumulate.
// Per block: packed-W chunk async-staged (ASYNCcnt) into double-buffered LDS,
// shared by all 8 waves. Per wave: 16-row x 128-col strip; A fragment and B
// fragment both register-double-buffered so every ds_load pair is hidden under
// a WMMA and the async stage is hidden under the whole K-step.
__global__ void gcn_gemm_bf16_wmma(const __bf16* __restrict__ Abf,
                                   const __bf16* __restrict__ Wp,
                                   float* __restrict__ out,
                                   int M, int K) {
    __shared__ __bf16 wlds[2][CHUNK];          // 16 KB double buffer
    const int tid  = threadIdx.x;
    const int wave = tid >> 5;
    const int lane = tid & 31;
    const int half = lane >> 4;
    const int lr   = lane & 15;
    const int m0   = (blockIdx.x * (blockDim.x >> 5) + wave) * 16;
    const bool active = (m0 < M);              // wave-uniform
    const int nkt = K >> 5;

    const __bf16* arow = Abf + (size_t)(active ? (m0 + lr) : 0) * K + half * 16;

    // stage chunk 0: each thread moves 32 contiguous bytes (2 async b128)
    {
        const __bf16* src = Wp + tid * 16;
        unsigned dst = (unsigned)(size_t)&wlds[0][tid * 16];
        asm volatile("global_load_async_to_lds_b128 %0, %1, off"
                     :: "v"(dst), "v"(src) : "memory");
        asm volatile("global_load_async_to_lds_b128 %0, %1, off offset:16"
                     :: "v"(dst), "v"(src) : "memory");
    }

    v8f acc[NT] = {};
    v16bf a_cur{};
    if (active) {
        v8bf alo = *(const v8bf*)(arow);
        v8bf ahi = *(const v8bf*)(arow + 8);
        a_cur = cat16(alo, ahi);
    }

    for (int kt = 0; kt < nkt; ++kt) {
        // my async stages complete -> barrier -> chunk kt visible to all waves
        asm volatile("s_wait_asynccnt 0" ::: "memory");
        __syncthreads();

        // stage chunk kt+1 into the other buffer (overlaps with WMMAs below)
        if (kt + 1 < nkt) {
            const __bf16* src = Wp + (size_t)(kt + 1) * CHUNK + tid * 16;
            unsigned dst = (unsigned)(size_t)&wlds[(kt + 1) & 1][tid * 16];
            asm volatile("global_load_async_to_lds_b128 %0, %1, off"
                         :: "v"(dst), "v"(src) : "memory");
            asm volatile("global_load_async_to_lds_b128 %0, %1, off offset:16"
                         :: "v"(dst), "v"(src) : "memory");
        }

        // prefetch next A fragment under the WMMA chain
        v16bf a_next{};
        if (active && kt + 1 < nkt) {
            v8bf alo = *(const v8bf*)(arow + (size_t)(kt + 1) * 32);
            v8bf ahi = *(const v8bf*)(arow + (size_t)(kt + 1) * 32 + 8);
            a_next = cat16(alo, ahi);
        }

        if (active) {
            const __bf16* lbase = &wlds[kt & 1][lane * 16];
            // B-fragment register double buffer: issue ds_loads for tile t+1
            // before the WMMA for tile t, hiding LDS latency under the XDL op.
            v8bf blo = *(const v8bf*)(lbase);
            v8bf bhi = *(const v8bf*)(lbase + 8);
            #pragma unroll
            for (int t = 0; t < NT; ++t) {
                v16bf b = cat16(blo, bhi);
                if (t + 1 < NT) {
                    const __bf16* bp = lbase + (t + 1) * (32 * 16);
                    blo = *(const v8bf*)(bp);
                    bhi = *(const v8bf*)(bp + 8);
                }
                acc[t] = __builtin_amdgcn_wmma_f32_16x16x32_bf16(
                    false, a_cur, false, b, (short)0, acc[t], false, false);
            }
        }
        a_cur = a_next;
    }

    if (active) {
        #pragma unroll
        for (int t = 0; t < NT; ++t)
            #pragma unroll
            for (int r = 0; r < 8; ++r)
                out[(size_t)(m0 + r + 8 * half) * HID + t * 16 + lr] = acc[t][r];
    }
}

__global__ void deg_init(float* __restrict__ deg) {
    int i = blockIdx.x * blockDim.x + threadIdx.x;
    if (i < N_NODES) deg[i] = 1.0f;            // self-loop weight
}

__global__ void deg_scatter(const long long* __restrict__ ei,
                            const float* __restrict__ w,
                            float* __restrict__ deg, int E) {
    int e = blockIdx.x * blockDim.x + threadIdx.x;
    if (e < E) atomicAdd(&deg[(int)ei[(size_t)E + e]], w[e]);
}

__global__ void deg_to_dinv(float* __restrict__ deg) {
    int i = blockIdx.x * blockDim.x + threadIdx.x;
    if (i < N_NODES) {
        float d = deg[i];
        deg[i] = (d > 0.0f) ? rsqrtf(d) : 0.0f;
    }
}

// h[i,f] = xw[i,f] * dinv[i]^2  (self-loop term, initializes the accumulator)
__global__ void scatter_init(const float* __restrict__ xw,
                             const float* __restrict__ dinv,
                             float* __restrict__ h) {
    int i = blockIdx.x * blockDim.x + threadIdx.x;
    if (i < N_NODES * HID) {
        float di = dinv[i >> 7];
        h[i] = xw[i] * di * di;
    }
}

// wave per edge: h[col] += dinv[row]*w*dinv[col] * xw[row]  (coalesced, 4 atomics/lane)
__global__ void edge_scatter(const long long* __restrict__ ei,
                             const float* __restrict__ w,
                             const float* __restrict__ dinv,
                             const float* __restrict__ xw,
                             float* __restrict__ h, int E) {
    int e = blockIdx.x * (blockDim.x >> 5) + (threadIdx.x >> 5);
    if (e >= E) return;
    int lane = threadIdx.x & 31;
    int r = (int)ei[e];
    int c = (int)ei[(size_t)E + e];
    float nrm = dinv[r] * w[e] * dinv[c];
    const float* src = xw + (size_t)r * HID;
    float*       dst = h  + (size_t)c * HID;
    #pragma unroll
    for (int j = 0; j < 4; ++j) {
        int f = lane + 32 * j;
        atomicAdd(&dst[f], nrm * src[f]);
    }
}

__global__ void bias_act(float* __restrict__ h, const float* __restrict__ b, int relu) {
    int i = blockIdx.x * blockDim.x + threadIdx.x;
    if (i < N_NODES * HID) {
        float v = h[i] + b[i & (HID - 1)];
        h[i] = relu ? fmaxf(v, 0.0f) : v;
    }
}

// wave per edge: out[e,:] = [h[row], h[col]] @ Wc + bc   (Wc: [2*HID, 2] row-major)
__global__ void edge_classifier(const long long* __restrict__ ei,
                                const float* __restrict__ h,
                                const float* __restrict__ Wc,
                                const float* __restrict__ bc,
                                float* __restrict__ out, int E) {
    int e = blockIdx.x * (blockDim.x >> 5) + (threadIdx.x >> 5);
    if (e >= E) return;
    int lane = threadIdx.x & 31;
    int r = (int)ei[e];
    int c = (int)ei[(size_t)E + e];
    const float* hr = h + (size_t)r * HID;
    const float* hc = h + (size_t)c * HID;
    float a0 = 0.0f, a1 = 0.0f;
    #pragma unroll
    for (int j = 0; j < 4; ++j) {
        int f = lane * 4 + j;
        float xr = hr[f], xc = hc[f];
        a0 += xr * Wc[f * 2 + 0] + xc * Wc[(HID + f) * 2 + 0];
        a1 += xr * Wc[f * 2 + 1] + xc * Wc[(HID + f) * 2 + 1];
    }
    for (int off = 16; off; off >>= 1) {
        a0 += __shfl_down(a0, off, 32);
        a1 += __shfl_down(a1, off, 32);
    }
    if (lane == 0) {
        out[(size_t)e * 2 + 0] = a0 + bc[0];
        out[(size_t)e * 2 + 1] = a1 + bc[1];
    }
}

extern "C" void kernel_launch(void* const* d_in, const int* in_sizes, int n_in,
                              void* d_out, int out_size, void* d_ws, size_t ws_size,
                              hipStream_t stream) {
    const float*     x  = (const float*)d_in[0];
    const long long* ei = (const long long*)d_in[1];   // int64 [2,E]
    const float*     w  = (const float*)d_in[2];
    const float*     W1 = (const float*)d_in[3];
    const float*     b1 = (const float*)d_in[4];
    const float*     W2 = (const float*)d_in[5];
    const float*     b2 = (const float*)d_in[6];
    const float*     Wc = (const float*)d_in[7];
    const float*     bc = (const float*)d_in[8];
    float* out = (float*)d_out;
    const int E = in_sizes[2];

    // workspace layout (all 16B-aligned): ~77 MB
    float*  bufA  = (float*)d_ws;                          // [N,HID] xw
    float*  bufB  = bufA + (size_t)N_NODES * HID;          // [N,HID] h
    float*  dinv  = bufB + (size_t)N_NODES * HID;          // [N] deg -> dinv
    __bf16* actbf = (__bf16*)(dinv + N_NODES);             // [N,IN_C] bf16 (x, then h)
    __bf16* w1p   = actbf + (size_t)N_NODES * IN_C;        // packed W1
    __bf16* w2p   = w1p + (size_t)IN_C * HID;              // packed W2

    const int TPB = 256;
    const int nblk_node = (N_NODES + TPB - 1) / TPB;
    const int nblk_e    = (E + TPB - 1) / TPB;
    const int nblk_nh   = (N_NODES * HID + TPB - 1) / TPB;
    const int nblk_gemm = (N_NODES / 16 + 7) / 8;          // 8 waves/block, 16 rows/wave
    const int nblk_wpe  = (E + 7) / 8;                     // wave-per-edge kernels

    // degree normalization
    deg_init   <<<nblk_node, TPB, 0, stream>>>(dinv);
    deg_scatter<<<nblk_e,    TPB, 0, stream>>>(ei, w, dinv, E);
    deg_to_dinv<<<nblk_node, TPB, 0, stream>>>(dinv);

    // weight packing (tiny)
    pack_weight<<<(IN_C * HID + TPB - 1) / TPB, TPB, 0, stream>>>(W1, w1p, IN_C);
    pack_weight<<<(HID * HID + TPB - 1) / TPB, TPB, 0, stream>>>(W2, w2p, HID);

    // layer 1: xw = x @ W1 ; h = relu(Â xw + b1)
    cvt_bf16<<<(N_NODES * IN_C + TPB - 1) / TPB, TPB, 0, stream>>>(x, actbf, N_NODES * IN_C);
    gcn_gemm_bf16_wmma<<<nblk_gemm, TPB, 0, stream>>>(actbf, w1p, bufA, N_NODES, IN_C);
    scatter_init      <<<nblk_nh,   TPB, 0, stream>>>(bufA, dinv, bufB);
    edge_scatter      <<<nblk_wpe,  TPB, 0, stream>>>(ei, w, dinv, bufA, bufB, E);
    bias_act          <<<nblk_nh,   TPB, 0, stream>>>(bufB, b1, 1);

    // layer 2: xw = h @ W2 ; h = Â xw + b2
    cvt_bf16<<<(N_NODES * HID + TPB - 1) / TPB, TPB, 0, stream>>>(bufB, actbf, N_NODES * HID);
    gcn_gemm_bf16_wmma<<<nblk_gemm, TPB, 0, stream>>>(actbf, w2p, bufA, N_NODES, HID);
    scatter_init      <<<nblk_nh,   TPB, 0, stream>>>(bufA, dinv, bufB);
    edge_scatter      <<<nblk_wpe,  TPB, 0, stream>>>(ei, w, dinv, bufA, bufB, E);
    bias_act          <<<nblk_nh,   TPB, 0, stream>>>(bufB, b2, 0);

    // edge classifier
    edge_classifier<<<nblk_wpe, TPB, 0, stream>>>(ei, bufB, Wc, bc, out, E);
}
